// CConv_33294586478861
// MI455X (gfx1250) — matrix operand order repaired and checked by reading, not compiled, not run on hardware
//
#include <hip/hip_runtime.h>
#include <hip/hip_bf16.h>

typedef __attribute__((ext_vector_type(16))) _Float16     v16h;
typedef __attribute__((ext_vector_type(8)))  float        v8f;
typedef __attribute__((ext_vector_type(4)))  unsigned int v4u;

#define BATCH    4
#define SEQ      4096
#define DIM      1024
#define CLEN     512
#define TILE_T   256     // outputs per channel-tile (one 16x16 f32 WMMA accumulator)
#define CHW      32      // channels per workgroup (one wave each)
#define NTHREADS 1024    // 32 waves (wave32)
#define NSTEP    17      // K-steps: kb = 32*s - 16 covers taps [m-16, m+526) >= [0,512)
#define XR_LEN   784     // time-reversed x slice: xin[t0-527 .. t0+256]
#define XR_STR   800     // f16 units per channel (padded)
#define F_STR    560     // f16 units per channel: filt_ext[-16 .. 543], zero padded
#define LDS_BYTES (2 * (CHW * XR_STR + 2 * CHW * F_STR))   // 122880 B < 320 KB

__global__ __launch_bounds__(NTHREADS)
void cconv_fir_wmma(const float* __restrict__ x,
                    const float* __restrict__ ctx,
                    const float* __restrict__ filt,
                    float* __restrict__ out) {
  extern __shared__ __align__(16) _Float16 smem[];
  _Float16* xrev  = smem;                   // [CHW][XR_STR]  xrev[ch][i] = xin[t0+256-i]
  _Float16* filtE = smem + CHW * XR_STR;    // [CHW][F_STR]   filtE[ch][i] = filt_ext[i-16]
  _Float16* filtO = filtE + CHW * F_STR;    // [CHW][F_STR]   filtO[ch][i] = filt_ext[i-15]

  const int tid = threadIdx.x;
  const int b   = blockIdx.z;
  const int t0  = blockIdx.y * TILE_T;
  const int d0  = blockIdx.x * CHW;

  // ---- zero both filter copies (covers the +/-16 zero pads) ----
  for (int idx = tid; idx < 2 * CHW * F_STR; idx += NTHREADS)
    filtE[idx] = (_Float16)0.f;
  __syncthreads();

  // ---- fill filter (coalesced: channel fastest, 128B rows) ----
  for (int idx = tid; idx < CHW * CLEN; idx += NTHREADS) {
    const int ch = idx & (CHW - 1);
    const int j  = idx >> 5;                       // tap
    const _Float16 h = (_Float16)filt[(size_t)j * DIM + d0 + ch];
    filtE[ch * F_STR + j + 16] = h;                // even-m aligned copy
    filtO[ch * F_STR + j + 15] = h;                // odd-m  aligned copy
  }

  // ---- fill time-reversed x slice incl. context, zero outside ----
  for (int idx = tid; idx < CHW * XR_LEN; idx += NTHREADS) {
    const int ch = idx & (CHW - 1);
    const int i  = idx >> 5;
    const int t  = t0 + TILE_T - i;                // t in [t0-527, t0+256]
    float v = 0.f;
    if (t >= 0 && t < SEQ)        v = x[((size_t)b * SEQ + t) * DIM + d0 + ch];
    else if (t >= -CLEN && t < 0) v = ctx[(size_t)(t + CLEN) * DIM + d0 + ch];
    xrev[ch * XR_STR + i] = (_Float16)v;
  }
  __syncthreads();

  // ---- per-wave WMMA K-loop: one channel, 256 outputs ----
  const int wave = tid >> 5;                       // channel within group
  const int lane = tid & 31;
  const int m    = lane & 15;                      // A row (== B column n)
  const int hi   = lane >> 4;                      // lane half
  const int modd = m & 1;

  const unsigned int* fb =
      (const unsigned int*)((modd ? filtO : filtE) + wave * F_STR);
  const unsigned int* xb = (const unsigned int*)(xrev + wave * XR_STR);

  v8f acc = {};
#pragma unroll
  for (int s = 0; s < NSTEP; ++s) {
    // A[m][k] = filt_ext[m + (32s-16) + k]; k order used here (contiguous, same as B):
    // VGPR j holds K = kk,kk+1 with kk = 2j + 16*hi
    union { unsigned int u[8]; v16h v; } A;
    const int eb = ((m - modd) + 32 * s) >> 1;     // aligned dword base
#pragma unroll
    for (int j = 0; j < 8; ++j) {
      const int kk = 2 * j + 16 * hi;
      A.u[j] = fb[eb + (kk >> 1)];
    }
    // B[k][n] = xrev[240 - 16n + 32s + k]; VGPR v holds K = 2v+16*hi (+1)
    // -> 8 contiguous, 32B-aligned dwords per lane: two b128 LDS loads
    union { v4u q[2]; v16h v; } Bm;
    const int rb = (240 - 16 * m + 32 * s + 16 * hi) >> 1;
    Bm.q[0] = *(const v4u*)(xb + rb);
    Bm.q[1] = *(const v4u*)(xb + rb + 4);

    acc = __builtin_amdgcn_wmma_f32_16x16x32_f16(
        false, A.v, false, Bm.v, (short)0, acc, false, false);
  }

  // ---- store: D layout -> lane(n=lane&15, half hi), VGPR r = M%8 ----
  // out time t = t0 + (r + 8*hi) + 16*n, channel d0+wave
  float* op = out + ((size_t)b * SEQ + t0 + 16 * m + 8 * hi) * DIM + (d0 + wave);
#pragma unroll
  for (int r = 0; r < 8; ++r)
    op[(size_t)r * DIM] = acc[r];
}

extern "C" void kernel_launch(void* const* d_in, const int* in_sizes, int n_in,
                              void* d_out, int out_size, void* d_ws, size_t ws_size,
                              hipStream_t stream) {
  const float* x    = (const float*)d_in[0];   // (4, 4096, 1024) f32
  const float* ctx  = (const float*)d_in[1];   // (512, 1024)     f32
  const float* filt = (const float*)d_in[2];   // (512, 1024)     f32
  float* out = (float*)d_out;                  // (4, 4096, 1024) f32

  dim3 grid(DIM / CHW, SEQ / TILE_T, BATCH);   // (32, 16, 4)
  dim3 block(NTHREADS);
  cconv_fir_wmma<<<grid, block, LDS_BYTES, stream>>>(x, ctx, filt, out);
}
